// CrossAttention_23072564314480
// MI455X (gfx1250) — matrix operand order repaired and checked by reading, not compiled, not run on hardware
//
#include <hip/hip_runtime.h>
#include <hip/hip_bf16.h>

typedef __attribute__((ext_vector_type(16))) _Float16 v16h;
typedef __attribute__((ext_vector_type(8)))  float    v8f;

#define B_    8
#define NT    1024          // query tokens (post-slice rows)
#define NK    1024          // key tokens
#define D_    768
#define H_    8
#define HD_   96
#define CLS_  20
#define SCALE_ 0.10206207261596575f   // 96^-0.5, folded into Q
#define NEG_   (-1e12f)
#define MASKR  0.3f
#define ATT_LD 1032                    // padded LDS stride (halves)

// ---------------- WMMA fragment helpers (wave32, 16x16x32 f16) ----------------
// A 16x32 f16, row-major source: lane<16 -> K 0-7 / 16-23 ; lane>=16 -> K 8-15 / 24-31
static __device__ __forceinline__ v16h ld_a16(const _Float16* src, int ld, int lane) {
  int row = lane & 15;
  int kb  = (lane & 16) ? 8 : 0;
  const _Float16* p = src + row * ld + kb;
  v16h a;
#pragma unroll
  for (int i = 0; i < 8; ++i) { a[i] = p[i]; a[8 + i] = p[16 + i]; }
  return a;
}
// B 32x16 where source is N-major (B[k][n] = src[n*ld + k]) : contiguous along K
static __device__ __forceinline__ v16h ld_bT16(const _Float16* src, int ld, int lane) {
  int col = lane & 15;
  int kb  = (lane & 16) ? 8 : 0;
  const _Float16* p = src + (size_t)col * ld + kb;
  v16h b;
#pragma unroll
  for (int e = 0; e < 8; ++e) { b[e] = p[e]; b[8 + e] = p[16 + e]; }
  return b;
}
static __device__ __forceinline__ v8f wmma_f16(v16h a, v16h b, v8f c) {
  return __builtin_amdgcn_wmma_f32_16x16x32_f16(false, a, false, b, (short)0, c, false, false);
}

// deterministic hash -> uniform [0,1)
static __device__ __forceinline__ float urand(unsigned x) {
  x ^= x >> 16; x *= 0x7feb352du; x ^= x >> 15; x *= 0x846ca68bu; x ^= x >> 16;
  return (float)(x >> 8) * (1.0f / 16777216.0f);
}

// ---------------- f32 -> f16 conversion ----------------
__global__ void cvt_f16(const float* __restrict__ in, _Float16* __restrict__ out, int n) {
  int i = blockIdx.x * 256 + threadIdx.x;
  if (i < n) out[i] = (_Float16)in[i];
}

// ---------------- f32 [K][N] -> f16 transposed [N][K] ----------------
__global__ void cvt_tr_f16(const float* __restrict__ in, _Float16* __restrict__ out,
                           int K, int N) {
  int k = blockIdx.x * 32 + (threadIdx.x & 31);
  int n = blockIdx.y * 8 + (threadIdx.x >> 5);
  if (k < K && n < N) out[(size_t)n * K + k] = (_Float16)in[(size_t)k * N + n];
}

// ============ generic inner body: 16-row slab GEMM, A staged in LDS ============
// Each wave computes 4 column tiles (64 cols); WT is N-major (K-contiguous).

// ---------------- Q = input_query @ Wq  (scaled, head layout) ----------------
__global__ __launch_bounds__(128) void gemm_q(const _Float16* __restrict__ A,
                                              const _Float16* __restrict__ WT,
                                              _Float16* __restrict__ qh) {
  __shared__ _Float16 As[16 * D_];                 // 24 KB
  const int l0 = blockIdx.x * 16;
  const int b  = blockIdx.z;
  const int wave = threadIdx.x >> 5;
  const int lane = threadIdx.x & 31;
  const int ctb  = (blockIdx.y * 4 + wave) * 4;    // 4 col tiles per wave

  const _Float16* Ab = A + ((size_t)b * NT + l0) * D_;
  for (int idx = threadIdx.x * 8; idx < 16 * D_; idx += 128 * 8)
    *(float4*)(&As[idx]) = *(const float4*)(Ab + idx);
  __syncthreads();

  v8f z = {};
  v8f acc[4];
#pragma unroll
  for (int j = 0; j < 4; ++j) acc[j] = z;

#pragma unroll 4
  for (int k0 = 0; k0 < D_; k0 += 32) {
    v16h af = ld_a16(&As[k0], D_, lane);
#pragma unroll
    for (int j = 0; j < 4; ++j) {
      v16h bf = ld_bT16(WT + (size_t)((ctb + j) * 16) * D_ + k0, D_, lane);
      acc[j] = wmma_f16(af, bf, acc[j]);
    }
  }

  const int coln = lane & 15, halfoff = (lane & 16) ? 8 : 0;
#pragma unroll
  for (int j = 0; j < 4; ++j) {
    const int c = (ctb + j) * 16 + coln;
    const int h = c / HD_, hd = c % HD_;
#pragma unroll
    for (int r = 0; r < 8; ++r) {
      int l = l0 + r + halfoff;
      qh[((size_t)(b * H_ + h) * NT + l) * HD_ + hd] = (_Float16)(acc[j][r] * SCALE_);
    }
  }
}

// ------- KV = input_key @ Wkv : K in head layout, V TRANSPOSED per head -------
__global__ __launch_bounds__(128) void gemm_kv(const _Float16* __restrict__ A,
                                               const _Float16* __restrict__ WT,  // [1536][768]
                                               _Float16* __restrict__ kh,        // [B,H,NK,HD]
                                               _Float16* __restrict__ vt) {      // [B,H,HD,NK]
  __shared__ _Float16 As[16 * D_];
  const int n0 = blockIdx.x * 16;
  const int b  = blockIdx.z;
  const int wave = threadIdx.x >> 5;
  const int lane = threadIdx.x & 31;
  const int ctb  = (blockIdx.y * 4 + wave) * 4;    // over 96 col tiles (1536 cols)

  const _Float16* Ab = A + ((size_t)b * NK + n0) * D_;
  for (int idx = threadIdx.x * 8; idx < 16 * D_; idx += 128 * 8)
    *(float4*)(&As[idx]) = *(const float4*)(Ab + idx);
  __syncthreads();

  v8f z = {};
  v8f acc[4];
#pragma unroll
  for (int j = 0; j < 4; ++j) acc[j] = z;

#pragma unroll 4
  for (int k0 = 0; k0 < D_; k0 += 32) {
    v16h af = ld_a16(&As[k0], D_, lane);
#pragma unroll
    for (int j = 0; j < 4; ++j) {
      v16h bf = ld_bT16(WT + (size_t)((ctb + j) * 16) * D_ + k0, D_, lane);
      acc[j] = wmma_f16(af, bf, acc[j]);
    }
  }

  const int coln = lane & 15, halfoff = (lane & 16) ? 8 : 0;
#pragma unroll
  for (int j = 0; j < 4; ++j) {
    int c = (ctb + j) * 16 + coln;
#pragma unroll
    for (int r = 0; r < 8; ++r) {
      int n = n0 + r + halfoff;
      _Float16 val = (_Float16)acc[j][r];
      if (c < D_) {
        int h = c / HD_, hd = c % HD_;
        kh[((size_t)(b * H_ + h) * NK + n) * HD_ + hd] = val;
      } else {
        int cc = c - D_;
        int h = cc / HD_, hd = cc % HD_;
        vt[((size_t)(b * H_ + h) * HD_ + hd) * NK + n] = val;
      }
    }
  }
}

// ---------------- cross-wave per-row reduction (rows split by half-wave) ----------------
static __device__ __forceinline__ void rowreduce(float v[8], int domax, float* red,
                                                 int wave, int lane) {
#pragma unroll
  for (int m = 1; m < 16; m <<= 1) {
#pragma unroll
    for (int r = 0; r < 8; ++r) {
      float o = __shfl_xor(v[r], m, 32);
      v[r] = domax ? fmaxf(v[r], o) : (v[r] + o);
    }
  }
  __syncthreads();
  if ((lane & 15) == 0) {
    int ro = (lane & 16) ? 8 : 0;
#pragma unroll
    for (int r = 0; r < 8; ++r) red[wave * 16 + ro + r] = v[r];
  }
  __syncthreads();
  int ro = (lane & 16) ? 8 : 0;
#pragma unroll
  for (int r = 0; r < 8; ++r) {
    float a0 = red[0 + ro + r],  a1 = red[16 + ro + r];
    float a2 = red[32 + ro + r], a3 = red[48 + ro + r];
    v[r] = domax ? fmaxf(fmaxf(a0, a1), fmaxf(a2, a3)) : (a0 + a1 + a2 + a3);
  }
}

// ---------------- fused attention: S=QK^T -> split softmax -> mask -> softmax -> PV ----
__global__ __launch_bounds__(128) void attn_kernel(const _Float16* __restrict__ qh,
                                                   const _Float16* __restrict__ kh,
                                                   const _Float16* __restrict__ vt,
                                                   _Float16* __restrict__ xh) {
  __shared__ _Float16 lds_attn[16 * ATT_LD];      // 33 KB: probabilities, f16
  __shared__ float    lds_xpart[4 * 16 * HD_];    // 24 KB: per-wave output partials
  __shared__ float    lds_red[64];                // cross-wave reduction scratch

  const int l0 = blockIdx.x * 16;
  const int h  = blockIdx.y;
  const int b  = blockIdx.z;
  const int wave = threadIdx.x >> 5;
  const int lane = threadIdx.x & 31;
  const int coln = lane & 15;
  const int halfoff = (lane & 16) ? 8 : 0;
  const int kstart = wave * 256;                  // this wave owns keys [kstart, kstart+256)

  const _Float16* qb  = qh + ((size_t)(b * H_ + h) * NT + l0) * HD_;
  const _Float16* kb_ = kh + (size_t)(b * H_ + h) * NK * HD_;
  const _Float16* vb  = vt + (size_t)(b * H_ + h) * HD_ * NK;   // [HD][NK]

  // --- Phase 1: scores, 16 rows x 256 keys per wave, kept in VGPR C-fragments ---
  v16h aq[3];
#pragma unroll
  for (int c = 0; c < 3; ++c) aq[c] = ld_a16(qb + c * 32, HD_, lane);

  v8f s[16];
#pragma unroll
  for (int t = 0; t < 16; ++t) {
    const _Float16* kt = kb_ + (size_t)(kstart + t * 16) * HD_;
    v8f acc = {};
#pragma unroll
    for (int c = 0; c < 3; ++c) {
      v16h bf = ld_bT16(kt + c * 32, HD_, lane);   // B = K^T, contiguous along head dim
      acc = wmma_f16(aq[c], bf, acc);
    }
    s[t] = acc;
  }

  // --- Phase 2a: split softmax (keys [0,20) vs [20,1024)) ---
  float mcls[8], mpat[8];
#pragma unroll
  for (int r = 0; r < 8; ++r) { mcls[r] = -1e30f; mpat[r] = -1e30f; }
#pragma unroll
  for (int t = 0; t < 16; ++t) {
    int key = kstart + t * 16 + coln;
    bool iscls = key < CLS_;
#pragma unroll
    for (int r = 0; r < 8; ++r) {
      float v = s[t][r];
      if (iscls) mcls[r] = fmaxf(mcls[r], v); else mpat[r] = fmaxf(mpat[r], v);
    }
  }
  rowreduce(mcls, 1, lds_red, wave, lane);
  rowreduce(mpat, 1, lds_red, wave, lane);

  float scls[8], spat[8];
#pragma unroll
  for (int r = 0; r < 8; ++r) { scls[r] = 0.f; spat[r] = 0.f; }
#pragma unroll
  for (int t = 0; t < 16; ++t) {
    int key = kstart + t * 16 + coln;
    bool iscls = key < CLS_;
#pragma unroll
    for (int r = 0; r < 8; ++r) {
      float e = __expf(s[t][r] - (iscls ? mcls[r] : mpat[r]));
      s[t][r] = e;
      if (iscls) scls[r] += e; else spat[r] += e;
    }
  }
  rowreduce(scls, 0, lds_red, wave, lane);
  rowreduce(spat, 0, lds_red, wave, lane);

  // --- Phase 2b: normalize, Bernoulli re-mask, second softmax over all keys ---
  float m2[8];
#pragma unroll
  for (int r = 0; r < 8; ++r) m2[r] = -1e30f;
#pragma unroll
  for (int t = 0; t < 16; ++t) {
    int key = kstart + t * 16 + coln;
    bool iscls = key < CLS_;
#pragma unroll
    for (int r = 0; r < 8; ++r) {
      float p = s[t][r] / (iscls ? scls[r] : spat[r]);
      unsigned lfull = (unsigned)(l0 + r + halfoff + CLS_);
      unsigned idx = ((unsigned)(b * H_ + h) * (unsigned)(CLS_ + NT) + lfull) * (unsigned)NK
                     + (unsigned)key;
      float v2 = p + ((urand(idx * 2654435761u + 0x9E3779B9u) < MASKR) ? NEG_ : 0.0f);
      s[t][r] = v2;
      m2[r] = fmaxf(m2[r], v2);
    }
  }
  rowreduce(m2, 1, lds_red, wave, lane);

  float s2[8];
#pragma unroll
  for (int r = 0; r < 8; ++r) s2[r] = 0.f;
#pragma unroll
  for (int t = 0; t < 16; ++t) {
#pragma unroll
    for (int r = 0; r < 8; ++r) {
      float e = __expf(s[t][r] - m2[r]);
      s[t][r] = e;
      s2[r] += e;
    }
  }
  rowreduce(s2, 0, lds_red, wave, lane);

  // write normalized probabilities to LDS as f16 (A-matrix staging for P*V)
#pragma unroll
  for (int t = 0; t < 16; ++t) {
    int key = kstart + t * 16 + coln;
#pragma unroll
    for (int r = 0; r < 8; ++r) {
      int row = r + halfoff;
      lds_attn[row * ATT_LD + key] = (_Float16)(s[t][r] / s2[r]);
    }
  }
  __syncthreads();

  // --- Phase 3: X = P @ V, each wave over its own 256-key slab, K chunks of 32 ---
  v8f z = {};
  v8f xacc[6];
#pragma unroll
  for (int ct = 0; ct < 6; ++ct) xacc[ct] = z;

#pragma unroll
  for (int cc = 0; cc < 8; ++cc) {
    int k0 = kstart + cc * 32;
    v16h af;
    {
      int row = lane & 15;
      int kb2 = (lane & 16) ? 8 : 0;
      const _Float16* p = &lds_attn[row * ATT_LD + k0 + kb2];
#pragma unroll
      for (int i = 0; i < 8; ++i) { af[i] = p[i]; af[8 + i] = p[16 + i]; }
    }
#pragma unroll
    for (int ct = 0; ct < 6; ++ct) {
      // V^T: B[k][n] = vb[(ct*16+col)*NK + k0 + k], contiguous along K
      v16h bf = ld_bT16(vb + (size_t)(ct * 16) * NK + k0, NK, lane);
      xacc[ct] = wmma_f16(af, bf, xacc[ct]);
    }
  }

  // per-wave partials -> LDS, deterministic 4-way tree sum, store f16 x
#pragma unroll
  for (int ct = 0; ct < 6; ++ct) {
#pragma unroll
    for (int r = 0; r < 8; ++r) {
      int row = r + halfoff;
      lds_xpart[(wave * 16 + row) * HD_ + ct * 16 + coln] = xacc[ct][r];
    }
  }
  __syncthreads();
  for (int i = threadIdx.x; i < 16 * HD_; i += 128) {
    int row = i / HD_, hd = i % HD_;
    float sum = (lds_xpart[(0 * 16 + row) * HD_ + hd] + lds_xpart[(1 * 16 + row) * HD_ + hd])
              + (lds_xpart[(2 * 16 + row) * HD_ + hd] + lds_xpart[(3 * 16 + row) * HD_ + hd]);
    xh[((size_t)b * NT + (l0 + row)) * D_ + h * HD_ + hd] = (_Float16)sum;
  }
}

// ---------------- out = X @ Wproj + bproj  (f32 output) ----------------
__global__ __launch_bounds__(128) void gemm_proj(const _Float16* __restrict__ A,
                                                 const _Float16* __restrict__ WT,
                                                 const float* __restrict__ bias,
                                                 float* __restrict__ out) {
  __shared__ _Float16 As[16 * D_];
  const int l0 = blockIdx.x * 16;
  const int b  = blockIdx.z;
  const int wave = threadIdx.x >> 5;
  const int lane = threadIdx.x & 31;
  const int ctb  = (blockIdx.y * 4 + wave) * 4;

  const _Float16* Ab = A + ((size_t)b * NT + l0) * D_;
  for (int idx = threadIdx.x * 8; idx < 16 * D_; idx += 128 * 8)
    *(float4*)(&As[idx]) = *(const float4*)(Ab + idx);
  __syncthreads();

  v8f z = {};
  v8f acc[4];
#pragma unroll
  for (int j = 0; j < 4; ++j) acc[j] = z;

#pragma unroll 4
  for (int k0 = 0; k0 < D_; k0 += 32) {
    v16h af = ld_a16(&As[k0], D_, lane);
#pragma unroll
    for (int j = 0; j < 4; ++j) {
      v16h bf = ld_bT16(WT + (size_t)((ctb + j) * 16) * D_ + k0, D_, lane);
      acc[j] = wmma_f16(af, bf, acc[j]);
    }
  }

  const int coln = lane & 15, halfoff = (lane & 16) ? 8 : 0;
#pragma unroll
  for (int j = 0; j < 4; ++j) {
    const int c = (ctb + j) * 16 + coln;
    const float bv = bias[c];
#pragma unroll
    for (int r = 0; r < 8; ++r) {
      int l = l0 + r + halfoff;
      out[((size_t)b * NT + l) * D_ + c] = acc[j][r] + bv;
    }
  }
}

// ---------------- host side ----------------
extern "C" void kernel_launch(void* const* d_in, const int* in_sizes, int n_in,
                              void* d_out, int out_size, void* d_ws, size_t ws_size,
                              hipStream_t stream) {
  (void)in_sizes; (void)n_in; (void)out_size; (void)ws_size;
  const size_t nIQ  = (size_t)B_ * NT * D_;
  const size_t nIK  = (size_t)B_ * NK * D_;
  const size_t nWq  = (size_t)D_ * D_;
  const size_t nWkv = (size_t)D_ * 2 * D_;
  const size_t nWp  = (size_t)D_ * D_;

  _Float16* hq    = (_Float16*)d_ws;         // f16 copies of activations
  _Float16* hk    = hq    + nIQ;
  _Float16* hWqT  = hk    + nIK;             // transposed f16 weights [N][K]
  _Float16* hWkvT = hWqT  + nWq;
  _Float16* hWpT  = hWkvT + nWkv;
  _Float16* qh    = hWpT  + nWp;             // [B,H,NT,HD] scaled Q
  _Float16* kh    = qh    + nIQ;             // [B,H,NK,HD]
  _Float16* vt    = kh    + nIK;             // [B,H,HD,NK]  (V transposed)
  _Float16* xh    = vt    + nIK;             // [B,NT,D] attention output

  const float* iq  = (const float*)d_in[0];
  const float* ik  = (const float*)d_in[1];
  const float* Wq  = (const float*)d_in[2];
  const float* Wkv = (const float*)d_in[3];
  // d_in[4] = Wcls: dead (output slice drops cls rows)
  const float* Wp  = (const float*)d_in[5];
  const float* bp  = (const float*)d_in[6];

  cvt_f16<<<dim3((unsigned)((nIQ + 255) / 256)), 256, 0, stream>>>(iq, hq, (int)nIQ);
  cvt_f16<<<dim3((unsigned)((nIK + 255) / 256)), 256, 0, stream>>>(ik, hk, (int)nIK);
  cvt_tr_f16<<<dim3(D_ / 32, D_ / 8),     256, 0, stream>>>(Wq,  hWqT,  D_, D_);
  cvt_tr_f16<<<dim3(D_ / 32, 2 * D_ / 8), 256, 0, stream>>>(Wkv, hWkvT, D_, 2 * D_);
  cvt_tr_f16<<<dim3(D_ / 32, D_ / 8),     256, 0, stream>>>(Wp,  hWpT,  D_, D_);

  gemm_q    <<<dim3(NT / 16, 3, B_), 128, 0, stream>>>(hq, hWqT,  qh);
  gemm_kv   <<<dim3(NK / 16, 6, B_), 128, 0, stream>>>(hk, hWkvT, kh, vt);
  attn_kernel<<<dim3(NT / 16, H_, B_), 128, 0, stream>>>(qh, kh, vt, xh);
  gemm_proj <<<dim3(NT / 16, 3, B_), 128, 0, stream>>>(xh, hWpT, bp, (float*)d_out);
}
